// ReviewAnalyzeModel_12738873000107
// MI455X (gfx1250) — matrix operand order repaired and checked by reading, not compiled, not run on hardware
//
#include <hip/hip_runtime.h>
#include <hip/hip_bf16.h>
#include <math.h>

// Problem constants (from reference): B=64, T=512, E=256, H=256, V=50000
#define B_   64
#define T_   512
#define E_   256
#define H_   256
#define G4H  1024          // 4*H
#define M0_  (T_ * B_)     // 32768 rows for the batched pre-GEMMs

// Recurrent kernel dynamic LDS: h (64x256 bf16 = 32 KB) + resident half of
// Whh (1024 rows x 128 K bf16 = 256 KB) = 288 KB of the WGP's 320 KB.
#define HLDS_ELEMS   (B_ * H_)          // 16384
#define WLDS_K       128                // K 0..127 resident in LDS
#define WLDS_ELEMS   (G4H * WLDS_K)     // 131072
#define REC_LDS_BYTES ((HLDS_ELEMS + WLDS_ELEMS) * 2)  // 294912 B

typedef __attribute__((ext_vector_type(16))) __bf16         bf16x16;
typedef __attribute__((ext_vector_type(8)))  float          f32x8;
typedef __attribute__((ext_vector_type(4)))  float          f32x4;
typedef __attribute__((ext_vector_type(16))) unsigned short u16x16;
typedef __attribute__((ext_vector_type(8)))  unsigned short u16x8;
typedef __attribute__((ext_vector_type(4)))  unsigned short u16x4;

__device__ __forceinline__ unsigned short f2bf(float f) {
  union { float f; unsigned u; } v; v.f = f;
  unsigned r = v.u + 0x7FFFu + ((v.u >> 16) & 1u);   // round-to-nearest-even
  return (unsigned short)(r >> 16);
}
__device__ __forceinline__ float sigmoidf_(float x) { return 1.0f / (1.0f + __expf(-x)); }

__device__ __forceinline__ f32x8 zero8() {
  f32x8 z;
#pragma unroll
  for (int i = 0; i < 8; ++i) z[i] = 0.0f;
  return z;
}

// Build a 16x32 bf16 A-fragment from two contiguous 8-element halves
// (lanes 0-15 hold K 0..15; lanes 16-31 hold K 16..31)
__device__ __forceinline__ bf16x16 make_afrag(const unsigned short* p) {
  u16x8 lo = *(const u16x8*)(p);
  u16x8 hi = *(const u16x8*)(p + 8);
  u16x16 v;
#pragma unroll
  for (int i = 0; i < 8; ++i) { v[i] = lo[i]; v[i + 8] = hi[i]; }
  return __builtin_bit_cast(bf16x16, v);
}

// ---------------------------------------------------------------------------
// Elementwise f32 -> bf16 conversion (weights)
// ---------------------------------------------------------------------------
__global__ void cvt_bf16_kernel(const float* __restrict__ src,
                                unsigned short* __restrict__ dst, int n) {
  int i = blockIdx.x * 256 + threadIdx.x;
  if (i < n) dst[i] = f2bf(src[i]);
}

// ---------------------------------------------------------------------------
// Embedding gather: xs[t*B+b, 4e..4e+3] = bf16(embed[x[b,t], 4e..4e+3])
// ---------------------------------------------------------------------------
__global__ void embed_gather_kernel(const int* __restrict__ x,
                                    const float* __restrict__ embed,
                                    unsigned short* __restrict__ xs) {
  int r = blockIdx.x;            // r = t*B + b
  int t = r >> 6;
  int b = r & 63;
  int tok = x[b * T_ + t];
  int e4 = threadIdx.x * 4;      // 64 threads cover E=256
  f32x4 v = *(const f32x4*)&embed[(size_t)tok * E_ + e4];
  u16x4 o;
#pragma unroll
  for (int i = 0; i < 4; ++i) o[i] = f2bf(v[i]);
  *(u16x4*)&xs[(size_t)r * E_ + e4] = o;
}

// ---------------------------------------------------------------------------
// Batched input projection: preG(frag layout) = A[M,K](bf16) @ W[1024,K]^T + bias
// One wave computes a 16(M) x 64(N) tile. grid = M0/8 blocks of 8 waves.
// ---------------------------------------------------------------------------
__global__ void __launch_bounds__(256)
pregemm_kernel(const unsigned short* __restrict__ A,
               const unsigned short* __restrict__ W,
               const float* __restrict__ bias,
               float* __restrict__ out, int K) {
  const int lane  = threadIdx.x & 31;
  const int wave  = threadIdx.x >> 5;
  const int gw    = blockIdx.x * 8 + wave;
  const int gmt   = gw >> 4;          // 2048 global M-tiles of 16 rows
  const int nt64  = gw & 15;          // 16 N-groups of 64 cols
  const int lhalf = lane >> 4;
  const int l16   = lane & 15;

  const int mrow = gmt * 16 + l16;
  const int aoff = lhalf * 16;

  f32x8 acc[4];
#pragma unroll
  for (int s = 0; s < 4; ++s) acc[s] = zero8();

  const int nchunks = K >> 5;
  for (int kc = 0; kc < nchunks; ++kc) {
    const unsigned short* ap = A + (size_t)mrow * K + kc * 32 + aoff;
    bf16x16 afrag = make_afrag(ap);
#pragma unroll
    for (int s = 0; s < 4; ++s) {
      int ncol = (nt64 * 4 + s) * 16 + l16;
      const unsigned short* bp = W + (size_t)ncol * K + kc * 32 + lhalf * 16;
      bf16x16 bfrag = __builtin_bit_cast(bf16x16, *(const u16x16*)bp);
      acc[s] = __builtin_amdgcn_wmma_f32_16x16x32_bf16(
          false, afrag, false, bfrag, (short)0, acc[s], false, false);
    }
  }

  // fragment-native store: one 32B f32x8 per tile per lane
  const int tt = gmt >> 2;        // timestep
  const int ml = gmt & 3;         // local batch tile
#pragma unroll
  for (int s = 0; s < 4; ++s) {
    int ntile = nt64 * 4 + s;
    float bv = bias[ntile * 16 + l16];
    f32x8 o = acc[s];
#pragma unroll
    for (int v = 0; v < 8; ++v) o[v] += bv;
    *(f32x8*)(out + (((size_t)tt * 256 + ntile * 4 + ml) << 8) + lane * 8) = o;
  }
}

// ---------------------------------------------------------------------------
// Recurrent LSTM chain. grid.x = 2 (dir 0 = forward, dir 1 = backward).
// 1024 threads = 32 waves; wave w: M-tile = w&3, hidden-col tile pair = (w>>2)*2.
// Each wave accumulates tiles for all 4 gates at the same hidden columns, so
// i/f/g/o for a cell live in the same lane+slot (lane-local nonlinearity).
// h lives in LDS (bf16, row-major = A-operand feed), c in VGPRs.
// K chunks 0-3 of Whh are LDS-resident (256 KB); chunks 4-7 stream from L2.
// preG is read in fragment-native layout (one f32x8 load per tile).
// ---------------------------------------------------------------------------
__global__ void __launch_bounds__(1024)
lstm_rec_kernel(const float* __restrict__ preG_f, const float* __restrict__ preG_b,
                const unsigned short* __restrict__ Whh_f,
                const unsigned short* __restrict__ Whh_b,
                unsigned short* __restrict__ hs_out,  // xs1 base [M0, 2H] bf16, or null
                float* __restrict__ hT_out,           // [2,64,256] f32, or null
                int write_hs) {
  extern __shared__ unsigned short smem[];
  unsigned short* h_lds = smem;                 // 64x256 bf16 = 32 KB
  unsigned short* w_lds = smem + HLDS_ELEMS;    // 1024x128 bf16 = 256 KB

  const int dir = blockIdx.x;
  const float* preG = dir ? preG_b : preG_f;
  const unsigned short* Whh = dir ? Whh_b : Whh_f;
  const int coloff = dir * H_;

  const int tid   = threadIdx.x;
  const int lane  = tid & 31;
  const int wave  = tid >> 5;
  const int mt    = wave & 3;           // batch-row tile
  const int jt0   = (wave >> 2) * 2;    // hidden-col tile pair (0..14 step 2)
  const int lhalf = lane >> 4;
  const int l16   = lane & 15;
  const int brow  = mt * 16 + l16;      // A-matrix row this lane feeds

  // cooperative copy indices (h_lds -> xs1), one 32B chunk per thread
  const int crow   = tid >> 4;          // 0..63
  const int cchunk = (tid & 15) * 16;   // 16 u16 = 32B chunks

  // zero h, stage resident half of Whh (K 0..127 for all 1024 rows) into LDS
  for (int i = tid; i < HLDS_ELEMS; i += 1024) h_lds[i] = 0;
  for (int i = tid; i < WLDS_ELEMS / 8; i += 1024) {
    int flat = i * 8;
    int n = flat >> 7;          // row
    int k = flat & 127;         // K within resident half
    *(u16x8*)&w_lds[flat] = *(const u16x8*)&Whh[(size_t)n * H_ + k];
  }
  __syncthreads();

  f32x8 cst[2], hlast[2];
#pragma unroll
  for (int j = 0; j < 2; ++j) { cst[j] = zero8(); hlast[j] = zero8(); }

  for (int s = 0; s < T_; ++s) {
    const int t  = dir ? (T_ - 1 - s) : s;
    const int rb = t * B_;

    // Prefetch next step's pre-gate fragments (global_prefetch_b8)
    if (s + 1 < T_) {
      const int tn = dir ? (T_ - 2 - s) : (s + 1);
#pragma unroll
      for (int j = 0; j < 2; ++j)
#pragma unroll
        for (int g = 0; g < 4; ++g) {
          int ntile = g * 16 + (jt0 + j);
          __builtin_prefetch(
              preG + (((size_t)tn * 256 + ntile * 4 + mt) << 8) + lane * 8, 0, 1);
        }
    }

    // Accumulators initialized from pre-computed x-projection (+bias),
    // fragment-native: one f32x8 (2x b128) load per tile
    f32x8 acc[2][4];
#pragma unroll
    for (int j = 0; j < 2; ++j) {
#pragma unroll
      for (int g = 0; g < 4; ++g) {
        int ntile = g * 16 + (jt0 + j);
        acc[j][g] = *(const f32x8*)(
            preG + (((size_t)t * 256 + ntile * 4 + mt) << 8) + lane * 8);
      }
    }

    // g += h @ Whh^T  (K = 256 in 8 chunks of 32; chunks 0-3 B from LDS,
    // chunks 4-7 B streamed from L2 -> overlapped DS + VMEM operand feed)
#pragma unroll
    for (int kc = 0; kc < 8; ++kc) {
      bf16x16 afrag = make_afrag(&h_lds[brow * H_ + kc * 32 + lhalf * 16]);
#pragma unroll
      for (int j = 0; j < 2; ++j) {
#pragma unroll
        for (int g = 0; g < 4; ++g) {
          int ncol = g * H_ + (jt0 + j) * 16 + l16;
          bf16x16 bfrag;
          if (kc < 4) {
            const unsigned short* bp =
                &w_lds[ncol * WLDS_K + kc * 32 + lhalf * 16];
            bfrag = __builtin_bit_cast(bf16x16, *(const u16x16*)bp);
          } else {
            const unsigned short* bp =
                Whh + (size_t)ncol * H_ + kc * 32 + lhalf * 16;
            bfrag = __builtin_bit_cast(bf16x16, *(const u16x16*)bp);
          }
          acc[j][g] = __builtin_amdgcn_wmma_f32_16x16x32_bf16(
              false, afrag, false, bfrag, (short)0, acc[j][g], false, false);
        }
      }
    }

    __syncthreads();  // all waves finished reading h_lds for this step

    // Gate nonlinearity + state update; write new h (bf16) to LDS
#pragma unroll
    for (int j = 0; j < 2; ++j) {
      int jcol = (jt0 + j) * 16 + l16;
#pragma unroll
      for (int v = 0; v < 8; ++v) {
        float iv = sigmoidf_(acc[j][0][v]);
        float fv = sigmoidf_(acc[j][1][v]);
        float gv = tanhf(acc[j][2][v]);
        float ov = sigmoidf_(acc[j][3][v]);
        float c  = fv * cst[j][v] + iv * gv;
        float h  = ov * tanhf(c);
        cst[j][v]   = c;
        hlast[j][v] = h;
        int bi = mt * 16 + lhalf * 8 + v;
        h_lds[bi * H_ + jcol] = f2bf(h);
      }
    }
    __syncthreads();  // new h complete before next step's A-fragment loads

    // Coalesced writeback of h to xs1: one 32B u16x16 store per thread
    if (write_hs) {
      u16x16 vv = *(const u16x16*)&h_lds[crow * H_ + cchunk];
      *(u16x16*)&hs_out[(size_t)(rb + crow) * (2 * H_) + coloff + cchunk] = vv;
    }
  }

  if (hT_out) {
#pragma unroll
    for (int j = 0; j < 2; ++j) {
      int jcol = (jt0 + j) * 16 + l16;
#pragma unroll
      for (int v = 0; v < 8; ++v) {
        int bi = mt * 16 + lhalf * 8 + v;
        hT_out[dir * (B_ * H_) + bi * H_ + jcol] = hlast[j][v];
      }
    }
  }
}

// ---------------------------------------------------------------------------
// Final head: out[b] = [hTf, hTb] . Wlin + blin
// ---------------------------------------------------------------------------
__global__ void __launch_bounds__(256)
linear_kernel(const float* __restrict__ hT, const float* __restrict__ Wlin,
              const float* __restrict__ blin, float* __restrict__ out) {
  __shared__ float red[256];
  int b = blockIdx.x, tid = threadIdx.x;
  float p = hT[b * H_ + tid] * Wlin[tid] +
            hT[B_ * H_ + b * H_ + tid] * Wlin[H_ + tid];
  red[tid] = p;
  __syncthreads();
  for (int off = 128; off > 0; off >>= 1) {
    if (tid < off) red[tid] += red[tid + off];
    __syncthreads();
  }
  if (tid == 0) out[b] = red[0] + blin[0];
}

// ---------------------------------------------------------------------------
extern "C" void kernel_launch(void* const* d_in, const int* in_sizes, int n_in,
                              void* d_out, int out_size, void* d_ws, size_t ws_size,
                              hipStream_t stream) {
  const int*   x        = (const int*)  d_in[0];
  const float* embed    = (const float*)d_in[1];
  const float* Wih_l0_f = (const float*)d_in[2];
  const float* Whh_l0_f = (const float*)d_in[3];
  const float* b_l0_f   = (const float*)d_in[4];
  const float* Wih_l0_b = (const float*)d_in[5];
  const float* Whh_l0_b = (const float*)d_in[6];
  const float* b_l0_b   = (const float*)d_in[7];
  const float* Wih_l1_f = (const float*)d_in[8];
  const float* Whh_l1_f = (const float*)d_in[9];
  const float* b_l1_f   = (const float*)d_in[10];
  const float* Wih_l1_b = (const float*)d_in[11];
  const float* Whh_l1_b = (const float*)d_in[12];
  const float* b_l1_b   = (const float*)d_in[13];
  const float* Wlin     = (const float*)d_in[14];
  const float* blin     = (const float*)d_in[15];
  float* out = (float*)d_out;

  // Workspace carve-up (~593 MB total)
  char*  ws  = (char*)d_ws;
  size_t off = 0;
  auto alloc = [&](size_t bytes) -> char* {
    char* p = ws + off;
    off += (bytes + 255) & ~(size_t)255;
    return p;
  };
  unsigned short* xs     = (unsigned short*)alloc((size_t)M0_ * E_ * 2);       // 16 MB
  unsigned short* wih0f  = (unsigned short*)alloc((size_t)G4H * E_ * 2);
  unsigned short* whh0f  = (unsigned short*)alloc((size_t)G4H * H_ * 2);
  unsigned short* wih0b  = (unsigned short*)alloc((size_t)G4H * E_ * 2);
  unsigned short* whh0b  = (unsigned short*)alloc((size_t)G4H * H_ * 2);
  unsigned short* wih1f  = (unsigned short*)alloc((size_t)G4H * 2 * H_ * 2);
  unsigned short* whh1f  = (unsigned short*)alloc((size_t)G4H * H_ * 2);
  unsigned short* wih1b  = (unsigned short*)alloc((size_t)G4H * 2 * H_ * 2);
  unsigned short* whh1b  = (unsigned short*)alloc((size_t)G4H * H_ * 2);
  unsigned short* xs1    = (unsigned short*)alloc((size_t)M0_ * 2 * H_ * 2);   // 33.5 MB
  float* preG0f = (float*)alloc((size_t)M0_ * G4H * 4);                        // 134 MB
  float* preG0b = (float*)alloc((size_t)M0_ * G4H * 4);                        // 134 MB
  float* preG1f = (float*)alloc((size_t)M0_ * G4H * 4);                        // 134 MB
  float* preG1b = (float*)alloc((size_t)M0_ * G4H * 4);                        // 134 MB
  float* hT     = (float*)alloc((size_t)2 * B_ * H_ * 4);

  // 1) weights -> bf16
  int nW0 = G4H * E_;       // 262144
  int nW1 = G4H * 2 * H_;   // 524288
  cvt_bf16_kernel<<<(nW0 + 255) / 256, 256, 0, stream>>>(Wih_l0_f, wih0f, nW0);
  cvt_bf16_kernel<<<(nW0 + 255) / 256, 256, 0, stream>>>(Whh_l0_f, whh0f, nW0);
  cvt_bf16_kernel<<<(nW0 + 255) / 256, 256, 0, stream>>>(Wih_l0_b, wih0b, nW0);
  cvt_bf16_kernel<<<(nW0 + 255) / 256, 256, 0, stream>>>(Whh_l0_b, whh0b, nW0);
  cvt_bf16_kernel<<<(nW1 + 255) / 256, 256, 0, stream>>>(Wih_l1_f, wih1f, nW1);
  cvt_bf16_kernel<<<(nW0 + 255) / 256, 256, 0, stream>>>(Whh_l1_f, whh1f, nW0);
  cvt_bf16_kernel<<<(nW1 + 255) / 256, 256, 0, stream>>>(Wih_l1_b, wih1b, nW1);
  cvt_bf16_kernel<<<(nW0 + 255) / 256, 256, 0, stream>>>(Whh_l1_b, whh1b, nW0);

  // 2) embedding gather (bf16 A-matrix for layer-0 pre-GEMM)
  embed_gather_kernel<<<M0_, 64, 0, stream>>>(x, embed, xs);

  // 3) layer-0 input projections (M=32768, N=1024, K=256) -> frag-layout preG
  pregemm_kernel<<<M0_ / 8, 256, 0, stream>>>(xs, wih0f, b_l0_f, preG0f, E_);
  pregemm_kernel<<<M0_ / 8, 256, 0, stream>>>(xs, wih0b, b_l0_b, preG0b, E_);

  // 4) layer-0 recurrence (both directions concurrently); writes xs1 = [hf|hb]
  lstm_rec_kernel<<<2, 1024, REC_LDS_BYTES, stream>>>(
      preG0f, preG0b, whh0f, whh0b, xs1, nullptr, 1);

  // 5) layer-1 input projections (K = 512)
  pregemm_kernel<<<M0_ / 8, 256, 0, stream>>>(xs1, wih1f, b_l1_f, preG1f, 2 * H_);
  pregemm_kernel<<<M0_ / 8, 256, 0, stream>>>(xs1, wih1b, b_l1_b, preG1b, 2 * H_);

  // 6) layer-1 recurrence; only final h needed
  lstm_rec_kernel<<<2, 1024, REC_LDS_BYTES, stream>>>(
      preG1f, preG1b, whh1f, whh1b, nullptr, hT, 0);

  // 7) head
  linear_kernel<<<B_, 256, 0, stream>>>(hT, Wlin, blin, out);

  (void)in_sizes; (void)n_in; (void)out_size; (void)ws_size;
}
